// CoreAttention_43353399885891
// MI455X (gfx1250) — matrix operand order repaired and checked by reading, not compiled, not run on hardware
//
#include <hip/hip_runtime.h>

// CDNA5 / gfx1250 flash attention forward (causal), fp32 in/out,
// f16 WMMA (v_wmma_f32_16x16x32_f16) with fp32 accumulation.
// Softmax runs in base-2 domain (log2e folded into the Q scale).
// Fragment loads are software-pipelined one step ahead of the WMMAs so
// ds_load latency hides under the matrix pipe (avoid s_wait_dscnt 0x0 chains).

typedef __attribute__((ext_vector_type(16))) _Float16 v16h;
typedef __attribute__((ext_vector_type(8)))  _Float16 h8;
typedef __attribute__((ext_vector_type(4)))  _Float16 h4;
typedef __attribute__((ext_vector_type(8)))  float    v8f;

#define BLOCK_M 128   // query rows per workgroup (16 per wave, 8 waves)
#define BLOCK_N 64    // key/value rows per kv tile
#define DHEAD   128
#define NWAVE   8
#define KSTR    136   // K LDS row stride (halves): 64 rows x 128 cols + pad
#define VSTR    72    // V^T LDS row stride (halves): 128 rows x 64 cols + pad
#define PSTR    72    // P LDS row stride (halves): 16 rows x 64 cols + pad

__device__ __forceinline__ float rmax16(float x) {
  // reduce over the 16 lanes of a half-wave (xor 1,2,4,8 stays in-half)
  x = fmaxf(x, __shfl_xor(x, 1, 32));
  x = fmaxf(x, __shfl_xor(x, 2, 32));
  x = fmaxf(x, __shfl_xor(x, 4, 32));
  x = fmaxf(x, __shfl_xor(x, 8, 32));
  return x;
}
__device__ __forceinline__ float rsum16(float x) {
  x += __shfl_xor(x, 1, 32);
  x += __shfl_xor(x, 2, 32);
  x += __shfl_xor(x, 4, 32);
  x += __shfl_xor(x, 8, 32);
  return x;
}

__device__ __forceinline__ v16h cat8(h8 lo, h8 hi) {
  return __builtin_shufflevector(lo, hi, 0, 1, 2, 3, 4, 5, 6, 7,
                                         8, 9, 10, 11, 12, 13, 14, 15);
}

__global__ __launch_bounds__(256, 1)
void fa_fwd_causal(const float* __restrict__ Q, const float* __restrict__ K,
                   const float* __restrict__ V, float* __restrict__ O, int S) {
  __shared__ _Float16 Klds[BLOCK_N][KSTR];        // K tile, row-major (key, d)
  __shared__ _Float16 Vt[DHEAD][VSTR];            // V tile, transposed (d, key)
  __shared__ _Float16 Plds[NWAVE][16][PSTR];      // per-wave P bounce buffer

  const int tid  = threadIdx.x;
  const int wave = tid >> 5;
  const int lane = tid & 31;
  const int l16  = lane & 15;
  const int hid  = lane >> 4;                     // half-wave id (0/1)

  const int q_tile = blockIdx.x * BLOCK_M;
  const long long base = (long long)blockIdx.y * S * DHEAD;   // (b*H+h) offset

  // 1/sqrt(128) * log2(e): scores come out in base-2 domain -> raw v_exp_f32
  const float scale = 0.08838834764831845f * 1.4426950408889634f;

  const int wrow0 = q_tile + wave * 16;           // first query row of wave
  const int wrow_last = wrow0 + 15;               // last query row of wave

  // ---- Q fragments: A-matrix 16x32 layout, 4 chunks cover D=128 ----
  // lane (l16) = row M; halves 0..7 -> K = c*32 + hid*16 + j,
  //                     halves 8..15 -> K = c*32 + hid*16 + 8 + j
  const float* qp = Q + base + (long long)(wrow0 + l16) * DHEAD;
  v16h qf[4];
  #pragma unroll
  for (int c = 0; c < 4; ++c) {
    union { v16h v; _Float16 e[16]; } u;
    const int d0 = c * 32 + hid * 16;
    #pragma unroll
    for (int j = 0; j < 8; ++j) u.e[j]     = (_Float16)(qp[d0 + j] * scale);
    #pragma unroll
    for (int j = 0; j < 8; ++j) u.e[8 + j] = (_Float16)(qp[d0 + 8 + j] * scale);
    qf[c] = u.v;
  }

  // ---- online-softmax state: row m = r + 8*hid, per-lane-half redundant ----
  float m_i[8], l_i[8];
  v8f o_acc[8];                                   // 16x128 output accumulator
  #pragma unroll
  for (int r = 0; r < 8; ++r) { m_i[r] = -1e30f; l_i[r] = 0.0f; }
  #pragma unroll
  for (int c = 0; c < 8; ++c) { v8f z = {}; o_acc[c] = z; }

  const int n_tiles = (q_tile + BLOCK_M) / BLOCK_N;   // causal upper bound
  for (int it = 0; it < n_tiles; ++it) {
    const int kv0 = it * BLOCK_N;

    // ---- stage K (f16 row-major) and V (f16 transposed) into LDS ----
    __syncthreads();
    {
      const float* kp = K + base + (long long)kv0 * DHEAD;
      const float* vp = V + base + (long long)kv0 * DHEAD;
      #pragma unroll
      for (int i = 0; i < 8; ++i) {
        const int idx = tid + i * 256;            // float4 index, 2048 total
        const int row = idx >> 5;                 // key row 0..63
        const int col = (idx & 31) << 2;          // d 0..124 step 4
        const float4 kq = *(const float4*)(kp + row * DHEAD + col);
        h4 kh; kh.x = (_Float16)kq.x; kh.y = (_Float16)kq.y;
               kh.z = (_Float16)kq.z; kh.w = (_Float16)kq.w;
        *(h4*)&Klds[row][col] = kh;
        const float4 vv = *(const float4*)(vp + row * DHEAD + col);
        Vt[col + 0][row] = (_Float16)vv.x;
        Vt[col + 1][row] = (_Float16)vv.y;
        Vt[col + 2][row] = (_Float16)vv.z;
        Vt[col + 3][row] = (_Float16)vv.w;
      }
    }
    __syncthreads();

    // prefetch next kv tile into L2 while we compute (global_prefetch_b8)
    if (it + 1 < n_tiles) {
      const long long nb = base + (long long)(kv0 + BLOCK_N) * DHEAD;
      __builtin_prefetch(K + nb + (long long)tid * 32, 0, 1);
      __builtin_prefetch(V + nb + (long long)tid * 32, 0, 1);
    }

    // Tiles entirely above the diagonal for this wave contribute nothing:
    // skip S / softmax / PV (wave-uniform branch; barriers stay outside).
    if (kv0 > wrow_last) continue;

    // ---- S = (Q*scale) @ K^T : 16 WMMAs, B-frags pipelined 1 ahead ----
    // step idx = t*4 + c : n-tile t (lane l16 = key col N), d-chunk c
    v8f s[4];
    #pragma unroll
    for (int t = 0; t < 4; ++t) { v8f z = {}; s[t] = z; }
    h8 blo, bhi;
    {
      const _Float16* kr0 = &Klds[l16][hid * 16];   // idx 0: t=0, c=0
      blo = *(const h8*)(kr0);
      bhi = *(const h8*)(kr0 + 8);
    }
    #pragma unroll
    for (int idx = 0; idx < 16; ++idx) {
      const int t = idx >> 2, c = idx & 3;
      h8 nlo = blo, nhi = bhi;
      if (idx < 15) {
        const int nt = (idx + 1) >> 2, nc = (idx + 1) & 3;
        const _Float16* kr = &Klds[nt * 16 + l16][nc * 32 + hid * 16];
        nlo = *(const h8*)(kr);
        nhi = *(const h8*)(kr + 8);
      }
      s[t] = __builtin_amdgcn_wmma_f32_16x16x32_f16(
          false, qf[c], false, cat8(blo, bhi), (short)0, s[t], false, false);
      blo = nlo; bhi = nhi;
    }

    // ---- causal mask (diagonal tiles only) + online softmax ----
    // wave-uniform: mask needed iff tile straddles this wave's diagonal
    const bool need_mask = (kv0 + BLOCK_N - 1) > wrow0;
    float p[4][8];
    #pragma unroll
    for (int r = 0; r < 8; ++r) {
      float tmax = -1e30f;
      if (need_mask) {
        const int mrow = wrow0 + r + 8 * hid;
        #pragma unroll
        for (int t = 0; t < 4; ++t) {
          const int ncol = kv0 + t * 16 + l16;
          const float sv = (ncol <= mrow) ? s[t][r] : -1e30f;
          s[t][r] = sv;
          tmax = fmaxf(tmax, sv);
        }
      } else {
        #pragma unroll
        for (int t = 0; t < 4; ++t) tmax = fmaxf(tmax, s[t][r]);
      }
      tmax = rmax16(tmax);
      const float mnew = fmaxf(m_i[r], tmax);
      const float corr = __builtin_amdgcn_exp2f(m_i[r] - mnew);
      float rs = 0.0f;
      #pragma unroll
      for (int t = 0; t < 4; ++t) {
        const float pv = __builtin_amdgcn_exp2f(s[t][r] - mnew);
        p[t][r] = pv;
        rs += pv;
      }
      rs = rsum16(rs);
      l_i[r] = l_i[r] * corr + rs;
      m_i[r] = mnew;
      #pragma unroll
      for (int c = 0; c < 8; ++c) o_acc[c][r] *= corr;
    }

    // First V fragment (ct=0, kk=0): depends only on staging, issue it
    // before the P bounce so it overlaps with softmax VALU + P stores.
    h8 vlo, vhi;
    {
      const _Float16* vr0 = &Vt[l16][hid * 16];
      vlo = *(const h8*)(vr0);
      vhi = *(const h8*)(vr0 + 8);
    }

    // ---- P: C-layout -> A-layout via wave-private LDS bounce ----
    #pragma unroll
    for (int t = 0; t < 4; ++t) {
      #pragma unroll
      for (int r = 0; r < 8; ++r)
        Plds[wave][r + 8 * hid][t * 16 + l16] = (_Float16)p[t][r];
    }
    v16h pa[2];
    {
      const _Float16* pr = &Plds[wave][l16][0];     // A: lane l16 = row M
      #pragma unroll
      for (int c = 0; c < 2; ++c) {
        const int kb = c * 32 + hid * 16;
        h8 lo = *(const h8*)(pr + kb);              // K = kb..kb+7
        h8 hi = *(const h8*)(pr + kb + 8);          // K = kb+8..kb+15
        pa[c] = cat8(lo, hi);
      }
    }

    // ---- O += P @ V : 16 WMMAs, B-frags pipelined 1 ahead ----
    // step idx = ct*2 + kk : d-tile ct (lane l16 = d col N), k-chunk kk
    #pragma unroll
    for (int idx = 0; idx < 16; ++idx) {
      const int ct = idx >> 1, kk = idx & 1;
      h8 nlo = vlo, nhi = vhi;
      if (idx < 15) {
        const int nct = (idx + 1) >> 1, nkk = (idx + 1) & 1;
        const _Float16* vr = &Vt[nct * 16 + l16][nkk * 32 + hid * 16];
        nlo = *(const h8*)(vr);
        nhi = *(const h8*)(vr + 8);
      }
      o_acc[ct] = __builtin_amdgcn_wmma_f32_16x16x32_f16(
          false, pa[kk], false, cat8(vlo, vhi), (short)0, o_acc[ct], false, false);
      vlo = nlo; vhi = nhi;
    }
  }

  // ---- normalize and write out (fp32) ----
  float* op = O + base + (long long)wrow0 * DHEAD;
  #pragma unroll
  for (int r = 0; r < 8; ++r) {
    const float inv = __builtin_amdgcn_rcpf(l_i[r]);
    const int m = r + 8 * hid;
    #pragma unroll
    for (int ct = 0; ct < 8; ++ct)
      op[(long long)m * DHEAD + ct * 16 + l16] = o_acc[ct][r] * inv;
  }
}

extern "C" void kernel_launch(void* const* d_in, const int* in_sizes, int n_in,
                              void* d_out, int out_size, void* d_ws, size_t ws_size,
                              hipStream_t stream) {
  (void)in_sizes; (void)n_in; (void)d_ws; (void)ws_size; (void)out_size;
  const float* Q = (const float*)d_in[0];
  const float* K = (const float*)d_in[1];
  const float* V = (const float*)d_in[2];
  float* O = (float*)d_out;
  const int B = 2, H = 16, S = 2048;
  dim3 grid(S / BLOCK_M, B * H);
  fa_fwd_causal<<<grid, dim3(256), 0, stream>>>(Q, K, V, O, S);
}
